// GATv2Architecture_4398046511397
// MI455X (gfx1250) — compile-verified, matrix operand order checked by reading, not run on hardware
//
#include <hip/hip_runtime.h>
#include <hip/hip_bf16.h>

// ---------------------------------------------------------------------------
// GATv2 (3 layers, 4/4/1 heads) + LN + 2 MLP heads, MI455X (gfx1250, wave32).
// Dense transforms use v_wmma_f32_16x16x32_bf16 (16x32 output strip per wave,
// A-fragment reused across two N-tiles); edge softmax uses L2-resident atomics
// (encoded-uint atomic max + f32 atomic add) with b128 gathers.
// ---------------------------------------------------------------------------

typedef __attribute__((ext_vector_type(16))) __bf16 v16bf;
typedef __attribute__((ext_vector_type(8)))  __bf16 v8bf;
typedef __attribute__((ext_vector_type(8)))  float  v8f;

union FragBF { v16bf v; v8bf h[2]; };

#define NEG_SLOPE 0.2f

// order-preserving float<->uint encoding for atomic segment-max
__device__ __forceinline__ unsigned enc_f32(float v) {
    unsigned u = __float_as_uint(v);
    return (u & 0x80000000u) ? ~u : (u | 0x80000000u);
}
__device__ __forceinline__ float dec_f32(unsigned e) {
    return (e & 0x80000000u) ? __uint_as_float(e & 0x7FFFFFFFu)
                             : __uint_as_float(~e);
}

__device__ __forceinline__ float lrelu(float v) {
    return (v > 0.f) ? v : NEG_SLOPE * v;
}

// ---------------------------------------------------------------------------
// zero fill (f32 / u32 share bit pattern 0)
// ---------------------------------------------------------------------------
__global__ void zero_kernel(float* p, long n) {
    long i = (long)blockIdx.x * blockDim.x + threadIdx.x;
    long stride = (long)gridDim.x * blockDim.x;
    for (; i < n; i += stride) p[i] = 0.0f;
}

// ---------------------------------------------------------------------------
// W [Kin,Kout] f32 (row major)  ->  Wt [Kout,Kin] bf16 (row major)
// ---------------------------------------------------------------------------
__global__ void cvt_w_kernel(const float* __restrict__ W, __bf16* __restrict__ Wt,
                             int Kin, int Kout) {
    int t = blockIdx.x * blockDim.x + threadIdx.x;
    if (t >= Kin * Kout) return;
    int ki = t / Kout, ko = t % Kout;
    Wt[ko * Kin + ki] = (__bf16)W[ki * Kout + ko];
}

// ---------------------------------------------------------------------------
// Fused LayerNorm: one wave per node, C in {64,128}.
//   x_c = (ssum ? acc[c]/ssum[head(c)] : xin[c])
//   y   = LN(x)*g + b (+ res_scale*res) (+ relu)
// writes f32 out and bf16 copy (next GEMM's A operand).
// ---------------------------------------------------------------------------
__global__ void ln_kernel(const float* __restrict__ xin,
                          const float* __restrict__ ssum, int H,
                          const float* __restrict__ g, const float* __restrict__ b,
                          const float* __restrict__ res, float res_scale,
                          int relu, int C, int Nn,
                          float* __restrict__ outf, __bf16* __restrict__ outb) {
    int wave = threadIdx.x >> 5;
    int lane = threadIdx.x & 31;
    int node = blockIdx.x * (blockDim.x >> 5) + wave;
    if (node >= Nn) return;

    int cpl = C >> 5;              // channels per lane (2 or 4)
    int hd  = C / (H > 0 ? H : 1); // head dim for the softmax divide
    float vals[4];
    float s = 0.f, s2 = 0.f;
    #pragma unroll
    for (int j = 0; j < 4; ++j) {
        if (j < cpl) {
            int c = lane + (j << 5);
            float v;
            if (ssum) {
                int h = c / hd;
                v = xin[(long)node * C + c] / ssum[(long)node * H + h];
            } else {
                v = xin[(long)node * C + c];
            }
            vals[j] = v;
            s  += v;
            s2 += v * v;
        }
    }
    #pragma unroll
    for (int m = 16; m >= 1; m >>= 1) {
        s  += __shfl_xor(s,  m, 32);
        s2 += __shfl_xor(s2, m, 32);
    }
    float mean = s / (float)C;
    float var  = s2 / (float)C - mean * mean;
    float inv  = rsqrtf(var + 1e-5f);

    #pragma unroll
    for (int j = 0; j < 4; ++j) {
        if (j < cpl) {
            int c = lane + (j << 5);
            float y = (vals[j] - mean) * inv * g[c] + b[c];
            if (res) y += res_scale * res[(long)node * C + c];
            if (relu) y = fmaxf(y, 0.f);
            outf[(long)node * C + c] = y;
            outb[(long)node * C + c] = (__bf16)y;
        }
    }
}

// ---------------------------------------------------------------------------
// WMMA GEMM: C[M,Nn] (f32) = A[M,K] (bf16, row major) x Bt[Nn,K]^T (bf16)
// One wave per 16x32 output strip (two 16x16 N-tiles sharing one A fragment:
// 2x v_wmma_f32_16x16x32_bf16 per K-step).  M mult of 16; Nn mult of 32;
// K mult of 32.
// ---------------------------------------------------------------------------
__global__ void gemm_bf16_wmma(const __bf16* __restrict__ A,
                               const __bf16* __restrict__ Bt,
                               const float* __restrict__ bias,
                               float* __restrict__ C,
                               int M, int K, int Nn, int relu) {
    int wave = threadIdx.x >> 5;
    int lane = threadIdx.x & 31;
    int strips_n = Nn >> 5;                 // 16x32 strips along N
    int strips   = (M >> 4) * strips_n;
    int strip = blockIdx.x * (blockDim.x >> 5) + wave;
    if (strip >= strips) return;            // wave-uniform: EXEC stays all-1s

    int tm = strip / strips_n;
    int tn = strip - tm * strips_n;
    int mbase = tm << 4, nbase = tn << 5;
    int lh  = lane >> 4;                    // lane half selects K sub-block
    int l15 = lane & 15;

    const __bf16* Arow  = A  + (long)(mbase + l15) * K;
    const __bf16* Brow0 = Bt + (long)(nbase + l15) * K;
    const __bf16* Brow1 = Bt + (long)(nbase + 16 + l15) * K;

    v8f acc0 = {}, acc1 = {};
    for (int k0 = 0; k0 < K; k0 += 32) {
        int kb = k0 + lh * 8;
        FragBF fa, fb0, fb1;
        fa.h[0]  = *(const v8bf*)(Arow  + kb);
        fa.h[1]  = *(const v8bf*)(Arow  + kb + 16);
        fb0.h[0] = *(const v8bf*)(Brow0 + kb);
        fb0.h[1] = *(const v8bf*)(Brow0 + kb + 16);
        fb1.h[0] = *(const v8bf*)(Brow1 + kb);
        fb1.h[1] = *(const v8bf*)(Brow1 + kb + 16);
        acc0 = __builtin_amdgcn_wmma_f32_16x16x32_bf16(
                   false, fa.v, false, fb0.v, (short)0, acc0, false, false);
        acc1 = __builtin_amdgcn_wmma_f32_16x16x32_bf16(
                   false, fa.v, false, fb1.v, (short)0, acc1, false, false);
    }

    float bv0 = bias ? bias[nbase + l15]      : 0.0f;
    float bv1 = bias ? bias[nbase + 16 + l15] : 0.0f;
    float* Cp0 = C + nbase + l15;
    float* Cp1 = C + nbase + 16 + l15;
    #pragma unroll
    for (int r = 0; r < 8; ++r) {
        int row = mbase + r + lh * 8;       // C/D layout: VGPR r -> M=r (+8 hi half)
        float v0 = acc0[r] + bv0;
        float v1 = acc1[r] + bv1;
        if (relu) { v0 = fmaxf(v0, 0.f); v1 = fmaxf(v1, 0.f); }
        Cp0[(long)row * Nn] = v0;
        Cp1[(long)row * Nn] = v1;
    }
}

// ---------------------------------------------------------------------------
// Edge pass A: per (edge, head) logit + encoded atomic segment-max over dst.
// features are 128-wide: H*HD == 128.  b128 gathers (rows 128B-aligned).
// ---------------------------------------------------------------------------
__global__ void edge_logits_kernel(const int* __restrict__ ei, int E, int Et,
                                   const float* __restrict__ xl,
                                   const float* __restrict__ xr,
                                   const float* __restrict__ att,
                                   float* __restrict__ logits,
                                   unsigned* __restrict__ mmax,
                                   int H, int HD) {
    int t = blockIdx.x * blockDim.x + threadIdx.x;
    if (t >= Et * H) return;
    int e = t / H, h = t - e * H;
    int s, d;
    if (e < E) { s = ei[e]; d = ei[E + e]; } else { s = d = e - E; }

    const float4* xs = (const float4*)(xl + (long)s * 128 + h * HD);
    const float4* xd = (const float4*)(xr + (long)d * 128 + h * HD);
    const float4* ah = (const float4*)(att + h * HD);
    __builtin_prefetch(xs, 0, 1);

    float acc = 0.f;
    int q = HD >> 2;
    for (int i = 0; i < q; ++i) {
        float4 a = xs[i];
        float4 c = xd[i];
        float4 w = ah[i];
        acc = fmaf(w.x, lrelu(a.x + c.x), acc);
        acc = fmaf(w.y, lrelu(a.y + c.y), acc);
        acc = fmaf(w.z, lrelu(a.z + c.z), acc);
        acc = fmaf(w.w, lrelu(a.w + c.w), acc);
    }
    logits[(long)e * H + h] = acc;
    atomicMax(&mmax[(long)d * H + h], enc_f32(acc));
}

// ---------------------------------------------------------------------------
// Edge pass B: one wave per edge.  p = exp(logit - max);
// acc[dst,c] += p * xl[src,c]; ssum[dst,h] += p (lane 0).
// ---------------------------------------------------------------------------
__global__ void edge_accum_kernel(const int* __restrict__ ei, int E, int Et,
                                  const float* __restrict__ xl,
                                  const float* __restrict__ logits,
                                  const unsigned* __restrict__ mmax,
                                  float* __restrict__ ssum,
                                  float* __restrict__ acc,
                                  int H) {
    int wave = threadIdx.x >> 5;
    int lane = threadIdx.x & 31;
    int e = blockIdx.x * (blockDim.x >> 5) + wave;
    if (e >= Et) return;
    int s, d;
    if (e < E) { s = ei[e]; d = ei[E + e]; } else { s = d = e - E; }

    const float* xs = xl + (long)s * 128;
    float*       ad = acc + (long)d * 128;
    __builtin_prefetch(xs, 0, 1);

    float p[4];
    #pragma unroll
    for (int j = 0; j < 4; ++j) {
        int c = lane + (j << 5);
        int h = (H == 4) ? j : 0;
        if (j == 0 || H == 4) {
            float lg = logits[(long)e * H + h];
            float mx = dec_f32(mmax[(long)d * H + h]);
            p[j] = __expf(lg - mx);
        } else {
            p[j] = p[0];
        }
        atomicAdd(&ad[c], p[j] * xs[c]);
    }
    if (lane == 0) {
        #pragma unroll
        for (int h = 0; h < 4; ++h)
            if (h < H) atomicAdd(&ssum[(long)d * H + h], p[h]);
    }
}

// ---------------------------------------------------------------------------
// Final heads: out[n,0] = t1[n,:]·w2a + b2a ; out[n,1] = t2[n,:]·w2b + b2b
// ---------------------------------------------------------------------------
__global__ void head_out_kernel(const float* __restrict__ t1,
                                const float* __restrict__ w2a, const float* __restrict__ b2a,
                                const float* __restrict__ t2,
                                const float* __restrict__ w2b, const float* __restrict__ b2b,
                                float* __restrict__ out, int Nn) {
    int n = blockIdx.x * blockDim.x + threadIdx.x;
    if (n >= Nn) return;
    float a = b2a[0], b = b2b[0];
    const float4* r1 = (const float4*)(t1 + (long)n * 64);
    const float4* r2 = (const float4*)(t2 + (long)n * 64);
    const float4* wa = (const float4*)w2a;
    const float4* wb = (const float4*)w2b;
    #pragma unroll
    for (int i = 0; i < 16; ++i) {
        float4 u = r1[i], v = r2[i], p = wa[i], q = wb[i];
        a = fmaf(u.x, p.x, fmaf(u.y, p.y, fmaf(u.z, p.z, fmaf(u.w, p.w, a))));
        b = fmaf(v.x, q.x, fmaf(v.y, q.y, fmaf(v.z, q.z, fmaf(v.w, q.w, b))));
    }
    out[(long)n * 2 + 0] = a;
    out[(long)n * 2 + 1] = b;
}

// ---------------------------------------------------------------------------
// host orchestration
// ---------------------------------------------------------------------------
extern "C" void kernel_launch(void* const* d_in, const int* in_sizes, int n_in,
                              void* d_out, int out_size, void* d_ws, size_t ws_size,
                              hipStream_t stream) {
    const int F_IN = 64, HID = 128;
    const int N  = in_sizes[0] / F_IN;
    const int E  = in_sizes[1] / 2;
    const int Et = E + N;

    const float* x      = (const float*)d_in[0];
    const int*   ei     = (const int*)  d_in[1];
    const float* ln_in_g = (const float*)d_in[2];
    const float* ln_in_b = (const float*)d_in[3];
    const float* w_l[3]  = { (const float*)d_in[4],  (const float*)d_in[9],  (const float*)d_in[14] };
    const float* w_r[3]  = { (const float*)d_in[5],  (const float*)d_in[10], (const float*)d_in[15] };
    const float* att[3]  = { (const float*)d_in[6],  (const float*)d_in[11], (const float*)d_in[16] };
    const float* ln_g[3] = { (const float*)d_in[7],  (const float*)d_in[12], (const float*)d_in[17] };
    const float* ln_b[3] = { (const float*)d_in[8],  (const float*)d_in[13], (const float*)d_in[18] };
    const float* rtt_w1 = (const float*)d_in[19]; const float* rtt_b1 = (const float*)d_in[20];
    const float* rtt_w2 = (const float*)d_in[21]; const float* rtt_b2 = (const float*)d_in[22];
    const float* ret_w1 = (const float*)d_in[23]; const float* ret_b1 = (const float*)d_in[24];
    const float* ret_w2 = (const float*)d_in[25]; const float* ret_b2 = (const float*)d_in[26];
    float* out = (float*)d_out;

    // ---- workspace arena ----
    char* wp = (char*)d_ws;
    auto take = [&](size_t bytes) -> void* {
        void* p = wp;
        wp += (bytes + 255) & ~(size_t)255;
        return p;
    };
    float*    hf     = (float*)   take((size_t)N * HID * 4);
    __bf16*   hb     = (__bf16*)  take((size_t)N * HID * 2);
    float*    xl     = (float*)   take((size_t)N * HID * 4);
    float*    xr     = (float*)   take((size_t)N * HID * 4);
    float*    accb   = (float*)   take((size_t)N * HID * 4);
    float*    logits = (float*)   take((size_t)Et * 4 * 4);
    unsigned* mmax   = (unsigned*)take((size_t)N * 4 * 4);
    float*    ssum   = (float*)   take((size_t)N * 4 * 4);
    __bf16*   wt     = (__bf16*)  take((size_t)HID * HID * 2);
    (void)ws_size;

    const int TB = 256;
    auto blk = [](long n, int t) { return (int)((n + t - 1) / t); };

    // 0) input LayerNorm (C=64) -> hf / hb
    ln_kernel<<<blk(N, 4), 128, 0, stream>>>(x, nullptr, 1, ln_in_g, ln_in_b,
                                             nullptr, 0.f, 0, F_IN, N, hf, hb);

    const int Kin[3]  = { F_IN, HID, HID };
    const int Hh[3]   = { 4, 4, 1 };
    const int HDh[3]  = { 32, 32, 128 };

    for (int L = 0; L < 3; ++L) {
        int K = Kin[L];
        int strips = (N >> 4) * (HID >> 5);  // 16x32 output strips per wave

        // x_l = h @ w_l
        cvt_w_kernel<<<blk((long)K * HID, TB), TB, 0, stream>>>(w_l[L], wt, K, HID);
        gemm_bf16_wmma<<<blk(strips, 4), 128, 0, stream>>>(hb, wt, nullptr, xl, N, K, HID, 0);
        // x_r = h @ w_r
        cvt_w_kernel<<<blk((long)K * HID, TB), TB, 0, stream>>>(w_r[L], wt, K, HID);
        gemm_bf16_wmma<<<blk(strips, 4), 128, 0, stream>>>(hb, wt, nullptr, xr, N, K, HID, 0);

        // segment-softmax buffers
        zero_kernel<<<2048, TB, 0, stream>>>(accb, (long)N * HID);
        zero_kernel<<<256,  TB, 0, stream>>>((float*)mmax, (long)N * Hh[L]);
        zero_kernel<<<256,  TB, 0, stream>>>(ssum, (long)N * Hh[L]);

        // edge passes
        edge_logits_kernel<<<blk((long)Et * Hh[L], TB), TB, 0, stream>>>(
            ei, E, Et, xl, xr, att[L], logits, mmax, Hh[L], HDh[L]);
        edge_accum_kernel<<<blk(Et, 4), 128, 0, stream>>>(
            ei, E, Et, xl, logits, mmax, ssum, accb, Hh[L]);

        // LN (+ divide by ssum) (+ 0.1*res for L>=1) (+ relu for L<2)
        const float* res = (L == 0) ? nullptr : hf;
        int relu = (L < 2) ? 1 : 0;
        ln_kernel<<<blk(N, 4), 128, 0, stream>>>(accb, ssum, Hh[L], ln_g[L], ln_b[L],
                                                 res, 0.1f, relu, HID, N, hf, hb);
    }

    // prediction heads: t = relu(h @ W1 + b1)  (WMMA, M=N, K=128, Nn=64)
    int strips64 = (N >> 4) * (64 >> 5);
    cvt_w_kernel<<<blk((long)HID * 64, TB), TB, 0, stream>>>(rtt_w1, wt, HID, 64);
    gemm_bf16_wmma<<<blk(strips64, 4), 128, 0, stream>>>(hb, wt, rtt_b1, xl, N, HID, 64, 1);
    cvt_w_kernel<<<blk((long)HID * 64, TB), TB, 0, stream>>>(ret_w1, wt, HID, 64);
    gemm_bf16_wmma<<<blk(strips64, 4), 128, 0, stream>>>(hb, wt, ret_b1, xr, N, HID, 64, 1);

    head_out_kernel<<<blk(N, TB), TB, 0, stream>>>(xl, rtt_w2, rtt_b2,
                                                   xr, ret_w2, ret_b2, out, N);
    (void)n_in; (void)out_size;
}
